// MixtureOfExperts_71330816852133
// MI455X (gfx1250) — matrix operand order repaired and checked by reading, not compiled, not run on hardware
//
#include <hip/hip_runtime.h>
#include <stdint.h>
#include <stddef.h>

// Problem constants (match reference: B=4, S=2048, D=512, F=2048, E=8, K=2)
#define TOK   8192      // B*S tokens
#define DDIM  512
#define FDIM  2048
#define ENUM  8

typedef __attribute__((ext_vector_type(16))) __bf16   v16bf;
typedef __attribute__((ext_vector_type(8)))  float    v8f;
typedef __attribute__((ext_vector_type(4)))  uint32_t v4u;
typedef __attribute__((ext_vector_type(8)))  uint32_t v8u;

// ---------- helpers ----------

static __device__ __forceinline__ uint16_t f2bf(float f) {
  union { float f; uint32_t u; } v; v.f = f;
  uint32_t r = v.u + 0x7FFFu + ((v.u >> 16) & 1u);   // round-to-nearest-even
  return (uint16_t)(r >> 16);
}

static __device__ __forceinline__ v16bf frag_from(v4u lo, v4u hi) {
  union { v8u u; v16bf b; } c;
  c.u = __builtin_shufflevector(lo, hi, 0, 1, 2, 3, 4, 5, 6, 7);
  return c.b;
}

// A fragment, 16x32 bf16, row-major source (LDS). Per ISA 7.12.2:
// lane<16 : row=lane,    k = kbase+{0..7} and kbase+{16..23}
// lane>=16: row=lane-16, k = kbase+{8..15} and kbase+{24..31}
// -> two contiguous 16-byte chunks per lane.
static __device__ __forceinline__ v16bf load_a_frag(const uint16_t* base, int stride,
                                                    int row, int half, int kbase) {
  const uint16_t* p = base + row * stride + kbase + half * 8;
  v4u lo = *(const v4u*)p;
  v4u hi = *(const v4u*)(p + 16);
  return frag_from(lo, hi);
}

// B fragment, 32x16 bf16, same k order as the A fragment (n=lane%16):
// lane<16: k = kbase+{0..7},{16..23}; lane>=16: k = kbase+{8..15},{24..31}.
// With K-major (transposed) weights: two 16-byte chunks per lane.
// `col` must already point at row n of the transposed weight matrix.
static __device__ __forceinline__ v16bf load_b_frag(const uint16_t* col, int half, int kbase) {
  const uint16_t* p = col + kbase + half * 8;
  v4u lo = *(const v4u*)p;
  v4u hi = *(const v4u*)(p + 16);
  return frag_from(lo, hi);
}

// ---------- small utility kernels ----------

__global__ void init_router_state_kernel(int* __restrict__ cnt, float* __restrict__ sumprob) {
  int i = threadIdx.x;
  if (i < ENUM) { cnt[i] = 0; sumprob[i] = 0.0f; }
}

__global__ void zero_f32_kernel(float* __restrict__ p, int n) {
  int i = blockIdx.x * blockDim.x + threadIdx.x;
  if (i < n) p[i] = 0.0f;
}

__global__ void cvt_bf16_kernel(const float* __restrict__ src, uint16_t* __restrict__ dst, int n) {
  int i = blockIdx.x * blockDim.x + threadIdx.x;
  if (i < n) dst[i] = f2bf(src[i]);
}

// W1 [E][D][F] fp32 -> w1t [E][F][D] bf16  (K-major for GEMM1 B fragments)
__global__ void transpose_w1_kernel(const float* __restrict__ W1, uint16_t* __restrict__ w1t) {
  int g = blockIdx.x * blockDim.x + threadIdx.x;
  if (g >= ENUM * DDIM * FDIM) return;
  int e = g / (DDIM * FDIM);
  int r = g % (DDIM * FDIM);
  int d = r / FDIM;
  int f = r % FDIM;
  w1t[((size_t)e * FDIM + f) * DDIM + d] = f2bf(W1[g]);
}

// W2 [E][F][D] fp32 -> w2t [E][D][F] bf16  (K-major for GEMM2 B fragments)
__global__ void transpose_w2_kernel(const float* __restrict__ W2, uint16_t* __restrict__ w2t) {
  int g = blockIdx.x * blockDim.x + threadIdx.x;
  if (g >= ENUM * FDIM * DDIM) return;
  int e = g / (FDIM * DDIM);
  int r = g % (FDIM * DDIM);
  int f = r / DDIM;
  int d = r % DDIM;
  w2t[((size_t)e * DDIM + d) * FDIM + f] = f2bf(W2[g]);
}

// ---------- router: logits -> softmax -> top-2 -> expert lists ----------

__global__ __launch_bounds__(256)
void router_kernel(const float* __restrict__ x, const float* __restrict__ Wg,
                   int* __restrict__ cnt, float* __restrict__ sumprob,
                   int* __restrict__ etok, float* __restrict__ ewt) {
  __shared__ float wgs[DDIM * ENUM];  // 16 KB
  int tid = threadIdx.x;
  for (int i = tid; i < DDIM * ENUM; i += 256) wgs[i] = Wg[i];
  __syncthreads();

  int t = blockIdx.x * 256 + tid;
  const float* xr = x + (size_t)t * DDIM;

  float acc[ENUM];
#pragma unroll
  for (int e = 0; e < ENUM; ++e) acc[e] = 0.0f;
  for (int d = 0; d < DDIM; ++d) {
    float xv = xr[d];
#pragma unroll
    for (int e = 0; e < ENUM; ++e) acc[e] += xv * wgs[d * ENUM + e];
  }

  float mx = acc[0];
#pragma unroll
  for (int e = 1; e < ENUM; ++e) mx = fmaxf(mx, acc[e]);
  float p[ENUM], s = 0.0f;
#pragma unroll
  for (int e = 0; e < ENUM; ++e) { p[e] = __expf(acc[e] - mx); s += p[e]; }
  float inv = 1.0f / s;
#pragma unroll
  for (int e = 0; e < ENUM; ++e) p[e] *= inv;

  // top-2 (ties -> lowest index, matching top_k)
  int i0 = 0;
#pragma unroll
  for (int e = 1; e < ENUM; ++e) if (p[e] > p[i0]) i0 = e;
  int i1 = (i0 == 0) ? 1 : 0;
#pragma unroll
  for (int e = 0; e < ENUM; ++e) if (e != i0 && p[e] > p[i1]) i1 = e;

  float wden = p[i0] + p[i1];
  float w0 = p[i0] / wden;
  float w1 = p[i1] / wden;

  int pos0 = atomicAdd(&cnt[i0], 1);
  etok[i0 * TOK + pos0] = t;
  ewt[i0 * TOK + pos0]  = w0;
  int pos1 = atomicAdd(&cnt[i1], 1);
  etok[i1 * TOK + pos1] = t;
  ewt[i1 * TOK + pos1]  = w1;

#pragma unroll
  for (int e = 0; e < ENUM; ++e) atomicAdd(&sumprob[e], p[e]);
}

// aux_loss = E * sum_e (cnt[e]/T) * (sumprob[e]/T), written at out[T*D]
__global__ void aux_loss_kernel(const int* __restrict__ cnt, const float* __restrict__ sumprob,
                                float* __restrict__ out) {
  if (blockIdx.x == 0 && threadIdx.x == 0) {
    float s = 0.0f;
    for (int e = 0; e < ENUM; ++e) s += (float)cnt[e] * sumprob[e];
    out[(size_t)TOK * DDIM] = (float)ENUM * s / ((float)TOK * (float)TOK);
  }
}

// ---------- main MoE FFN: gathered-token tiles, bf16 WMMA, f32 accumulate ----------
// grid = (TOK/16, ENUM), block = 256 threads (8 wave32s).
// Each block: 16 tokens of one expert, full D=512 output.
// F processed in slabs of 128: GEMM1 (each wave -> one 16x16 H tile, K=512),
// relu+bias -> LDS bf16, GEMM2 (each wave -> 4 output tiles, K=128).

__global__ __launch_bounds__(256)
void moe_ffn_kernel(const uint16_t* __restrict__ xb,
                    const uint16_t* __restrict__ w1t,   // [E][F][D] bf16
                    const uint16_t* __restrict__ w2t,   // [E][D][F] bf16
                    const float* __restrict__ b1,       // [E][F]
                    const float* __restrict__ b2,       // [E][D]
                    const int* __restrict__ cnt,
                    const int* __restrict__ etok,
                    const float* __restrict__ ewt,
                    float* __restrict__ out) {
  __shared__ __align__(16) uint16_t Xs[16][520];  // 16 tokens x 512 (pad 8 keeps 16B chunks)
  __shared__ __align__(16) uint16_t Hs[16][136];  // 16 tokens x 128 hidden slab
  __shared__ int   tok_s[16];
  __shared__ float wt_s[16];

  const int e    = blockIdx.y;
  const int base = blockIdx.x * 16;
  const int ne   = cnt[e];
  if (base >= ne) return;

  const int tid  = threadIdx.x;
  const int lane = tid & 31;
  const int wv   = tid >> 5;        // wave id 0..7
  const int half = lane >> 4;
  const int l15  = lane & 15;

  if (tid < 16) {
    int idx = base + tid;
    if (idx < ne) { tok_s[tid] = etok[e * TOK + idx]; wt_s[tid] = ewt[e * TOK + idx]; }
    else          { tok_s[tid] = etok[e * TOK + base]; wt_s[tid] = 0.0f; }
  }
  __syncthreads();

  { // stage X tile: 16 rows x 512 bf16; 16 threads/row x 32 elems (64B) each
    int row = tid >> 4;
    int c0  = (tid & 15) * 32;
    const v4u* src = (const v4u*)(xb + (size_t)tok_s[row] * DDIM + c0);
    v4u* dst = (v4u*)(&Xs[row][c0]);
    dst[0] = src[0]; dst[1] = src[1]; dst[2] = src[2]; dst[3] = src[3];
  }
  __syncthreads();

  v8f acc[4];
#pragma unroll
  for (int j = 0; j < 4; ++j) acc[j] = (v8f){0.f, 0.f, 0.f, 0.f, 0.f, 0.f, 0.f, 0.f};

  for (int f0 = 0; f0 < FDIM; f0 += 128) {
    // ---- GEMM1: wave wv computes H tile columns [f0+wv*16, f0+wv*16+16) ----
    v8f h = (v8f){0.f, 0.f, 0.f, 0.f, 0.f, 0.f, 0.f, 0.f};
    const uint16_t* w1col = w1t + ((size_t)e * FDIM + f0 + wv * 16 + l15) * DDIM;
#pragma unroll
    for (int k0 = 0; k0 < DDIM; k0 += 32) {
      v16bf a = load_a_frag(&Xs[0][0], 520, l15, half, k0);
      v16bf b = load_b_frag(w1col, half, k0);
      h = __builtin_amdgcn_wmma_f32_16x16x32_bf16(false, a, false, b, (short)0, h, false, false);
    }
    float b1v = b1[e * FDIM + f0 + wv * 16 + l15];

    __syncthreads();   // previous slab's Hs readers are done
#pragma unroll
    for (int i = 0; i < 8; ++i) {
      float v = h[i] + b1v;                       // C layout: row=i+8*half, col=l15
      Hs[i + 8 * half][wv * 16 + l15] = f2bf(v > 0.0f ? v : 0.0f);
    }
    __syncthreads();

    // ---- GEMM2: acc[j] covers out columns [(wv*4+j)*16, +16), K = this 128-slab ----
#pragma unroll
    for (int j = 0; j < 4; ++j) {
      const uint16_t* w2col =
          w2t + ((size_t)e * DDIM + (wv * 4 + j) * 16 + l15) * FDIM + f0;
#pragma unroll
      for (int k0 = 0; k0 < 128; k0 += 32) {
        v16bf a = load_a_frag(&Hs[0][0], 136, l15, half, k0);
        v16bf b = load_b_frag(w2col, half, k0);
        acc[j] = __builtin_amdgcn_wmma_f32_16x16x32_bf16(false, a, false, b, (short)0,
                                                         acc[j], false, false);
      }
    }
  }

  // ---- epilogue: + b2, scale by routing weight, combine into out ----
#pragma unroll
  for (int j = 0; j < 4; ++j) {
    int colb = (wv * 4 + j) * 16 + l15;
    float b2v = b2[e * DDIM + colb];
#pragma unroll
    for (int i = 0; i < 8; ++i) {
      int r = i + 8 * half;
      if (base + r < ne) {
        float v = (acc[j][i] + b2v) * wt_s[r];
        atomicAdd(&out[(size_t)tok_s[r] * DDIM + colb], v);
      }
    }
  }
}

// ---------- host launcher ----------

extern "C" void kernel_launch(void* const* d_in, const int* in_sizes, int n_in,
                              void* d_out, int out_size, void* d_ws, size_t ws_size,
                              hipStream_t stream) {
  (void)in_sizes; (void)n_in; (void)out_size; (void)ws_size;

  const float* x  = (const float*)d_in[0];   // [T, D]
  const float* Wg = (const float*)d_in[1];   // [D, E]
  const float* W1 = (const float*)d_in[2];   // [E, D, F]
  const float* b1 = (const float*)d_in[3];   // [E, F]
  const float* W2 = (const float*)d_in[4];   // [E, F, D]
  const float* b2 = (const float*)d_in[5];   // [E, D]
  float* out = (float*)d_out;                // [T*D] + 1 (aux)

  // workspace layout
  char* ws = (char*)d_ws;
  size_t off = 0;
  int*      cnt     = (int*)(ws + off);      off += 256;
  float*    sumprob = (float*)(ws + off);    off += 256;
  int*      etok    = (int*)(ws + off);      off += (size_t)ENUM * TOK * sizeof(int);
  float*    ewt     = (float*)(ws + off);    off += (size_t)ENUM * TOK * sizeof(float);
  uint16_t* xb      = (uint16_t*)(ws + off); off += (size_t)TOK * DDIM * 2;
  uint16_t* w1t     = (uint16_t*)(ws + off); off += (size_t)ENUM * DDIM * FDIM * 2;
  uint16_t* w2t     = (uint16_t*)(ws + off); off += (size_t)ENUM * DDIM * FDIM * 2;

  const int nOut = TOK * DDIM;               // 4,194,304
  const int nW   = ENUM * DDIM * FDIM;       // 8,388,608

  init_router_state_kernel<<<1, 32, 0, stream>>>(cnt, sumprob);
  zero_f32_kernel<<<nOut / 256, 256, 0, stream>>>(out, nOut);
  cvt_bf16_kernel<<<nOut / 256, 256, 0, stream>>>(x, xb, nOut);
  transpose_w1_kernel<<<nW / 256, 256, 0, stream>>>(W1, w1t);
  transpose_w2_kernel<<<nW / 256, 256, 0, stream>>>(W2, w2t);
  router_kernel<<<TOK / 256, 256, 0, stream>>>(x, Wg, cnt, sumprob, etok, ewt);
  aux_loss_kernel<<<1, 32, 0, stream>>>(cnt, sumprob, out);
  moe_ffn_kernel<<<dim3(TOK / 16, ENUM), 256, 0, stream>>>(
      xb, w1t, w2t, b1, b2, cnt, etok, ewt, out);
}
